// DecoupledPointJAFAR_58119497449869
// MI455X (gfx1250) — matrix operand order repaired and checked by reading, not compiled, not run on hardware
//
#include <hip/hip_runtime.h>
#include <hip/hip_bf16.h>
#include <math.h>

// Problem constants (match reference)
#define BATCH 2
#define NHR   16384
#define MLR   4096
#define KNN   16
#define CH    64
#define GEOC  18
#define EPSBN 1e-5f

typedef __attribute__((ext_vector_type(16))) _Float16 v16h;
typedef __attribute__((ext_vector_type(8)))  float    v8f;

// ------------------------------------------------------------------
// zero scratch stats
// ------------------------------------------------------------------
__global__ void zero_kernel(float* __restrict__ p, int n) {
  int i = blockIdx.x * blockDim.x + threadIdx.x;
  if (i < n) p[i] = 0.f;
}

// ------------------------------------------------------------------
// generic small 1x1 conv (VALU): out[b,co,p] = W[co,:] . x[b,:,p] + bias
// optional per-channel sum/sumsq accumulation for BatchNorm
// ------------------------------------------------------------------
template<int CIN, int COUT, bool STATS>
__global__ void conv_small_kernel(const float* __restrict__ x, const float* __restrict__ W,
                                  const float* __restrict__ bias, float* __restrict__ out,
                                  int P, float* __restrict__ stat) {
  __shared__ float sW[COUT * CIN];
  __shared__ float sB[COUT];
  __shared__ float sSum[COUT];
  __shared__ float sSq[COUT];
  const int b = blockIdx.y;
  const float* xb = x + (size_t)b * CIN * P;
  float* ob = out + (size_t)b * COUT * P;
  for (int i = threadIdx.x; i < COUT * CIN; i += blockDim.x) sW[i] = W[i];
  for (int i = threadIdx.x; i < COUT; i += blockDim.x) {
    sB[i] = bias[i];
    if (STATS) { sSum[i] = 0.f; sSq[i] = 0.f; }
  }
  __syncthreads();
  const int p = blockIdx.x * blockDim.x + threadIdx.x;
  const bool ok = (p < P);
  float xi[CIN];
  #pragma unroll
  for (int c = 0; c < CIN; ++c) xi[c] = ok ? xb[c * P + p] : 0.f;
  const int lane = threadIdx.x & 31;
  for (int co = 0; co < COUT; ++co) {
    float acc = sB[co];
    #pragma unroll
    for (int c = 0; c < CIN; ++c) acc = fmaf(sW[co * CIN + c], xi[c], acc);
    if (ok) ob[co * P + p] = acc;
    if (STATS) {
      float v = ok ? acc : 0.f;
      float s2 = v * v;
      #pragma unroll
      for (int off = 16; off > 0; off >>= 1) {
        v  += __shfl_xor(v, off);
        s2 += __shfl_xor(s2, off);
      }
      if (lane == 0) { atomicAdd(&sSum[co], v); atomicAdd(&sSq[co], s2); }
    }
  }
  if (STATS) {
    __syncthreads();
    for (int i = threadIdx.x; i < COUT; i += blockDim.x) {
      atomicAdd(&stat[i], sSum[i]);
      atomicAdd(&stat[COUT + i], sSq[i]);
    }
  }
}

// ------------------------------------------------------------------
// 64->64 1x1 conv via V_WMMA_F32_16X16X32_F16.
// Block = 256 threads (8 waves), processes 128 points of one batch.
// Activations staged point-major (padded stride 72) so each B fragment
// is 2 contiguous ds_load_b128; weights staged row-major f16 so each
// A fragment half is 1 ds_load_b128.
// ------------------------------------------------------------------
#define XPAD 72
__global__ void conv64_wmma_kernel(const float* __restrict__ x, const float* __restrict__ W,
                                   const float* __restrict__ bias, float* __restrict__ out,
                                   int P, float* __restrict__ stat) {
  __shared__ _Float16 sW[64 * 64];
  __shared__ _Float16 sXT[128 * XPAD]; // [point][channel], padded
  __shared__ float sSum[64];
  __shared__ float sSq[64];
  const int b  = blockIdx.y;
  const int pb = blockIdx.x * 128;
  const float* xb = x + (size_t)b * 64 * P;
  float* ob = out + (size_t)b * 64 * P;
  const int tid = threadIdx.x;

  for (int i = tid; i < 64 * 64; i += 256) sW[i] = (_Float16)W[i];
  for (int i = tid; i < 64 * 128; i += 256) {
    int ci = i >> 7, p = i & 127;
    int gp = pb + p;
    float v = (gp < P) ? xb[(size_t)ci * P + gp] : 0.f;
    sXT[p * XPAD + ci] = (_Float16)v;
  }
  if (stat && tid < 64) { sSum[tid] = 0.f; sSq[tid] = 0.f; }
  __syncthreads();

  const int wave = tid >> 5;
  const int lane = tid & 31;
  const int lcol = lane & 15;   // column (point) / row-in-tile index
  const int lhi  = lane >> 4;   // lane half
  const int colBase = wave * 16;
  const int myp = (colBase + lcol) * XPAD;

  v8f acc[4] = {};
  #pragma unroll
  for (int ks = 0; ks < 2; ++ks) {
    const int k0 = ks * 32;
    // B fragment (32x16): col = colBase+lcol, K = lhi*16 + h  (contiguous halfs)
    v16h bx;
    #pragma unroll
    for (int h = 0; h < 16; ++h)
      bx[h] = sXT[myp + k0 + lhi * 16 + h];
    #pragma unroll
    for (int mt = 0; mt < 4; ++mt) {
      const int m = mt * 16 + lcol; // A row for this lane
      v16h ax;
      #pragma unroll
      for (int h = 0; h < 8; ++h)
        ax[h] = sW[m * 64 + k0 + lhi * 16 + h];
      #pragma unroll
      for (int h = 0; h < 8; ++h)
        ax[8 + h] = sW[m * 64 + k0 + lhi * 16 + 8 + h];
      acc[mt] = __builtin_amdgcn_wmma_f32_16x16x32_f16(
          false, ax, false, bx, (short)0, acc[mt], false, false);
    }
  }

  // C/D layout: lane half lhi -> rows r+8*lhi within tile, col = lcol
  const int gcol = pb + colBase + lcol;
  const bool okc = (gcol < P);
  #pragma unroll
  for (int mt = 0; mt < 4; ++mt) {
    #pragma unroll
    for (int r = 0; r < 8; ++r) {
      const int m = mt * 16 + lhi * 8 + r;
      float v = acc[mt][r] + bias[m];
      if (okc) ob[(size_t)m * P + gcol] = v;
      if (stat) {
        float sv = okc ? v : 0.f;
        float s2 = sv * sv;
        #pragma unroll
        for (int off = 1; off < 16; off <<= 1) {
          sv += __shfl_xor(sv, off);
          s2 += __shfl_xor(s2, off);
        }
        if (lcol == 0) { atomicAdd(&sSum[m], sv); atomicAdd(&sSq[m], s2); }
      }
    }
  }
  if (stat) {
    __syncthreads();
    for (int i = tid; i < 64; i += 256) {
      atomicAdd(&stat[i], sSum[i]);
      atomicAdd(&stat[64 + i], sSq[i]);
    }
  }
}

// ------------------------------------------------------------------
// finalize BN stats: stat[0:C]=sum -> mean, stat[C:2C]=sumsq -> invstd
// ------------------------------------------------------------------
__global__ void bn_finalize_kernel(float* __restrict__ stat, int C, float cnt) {
  int c = threadIdx.x;
  if (c < C) {
    float mean = stat[c] / cnt;
    float var  = stat[C + c] / cnt - mean * mean;
    stat[c]     = mean;
    stat[C + c] = rsqrtf(var + EPSBN);
  }
}

// ------------------------------------------------------------------
// in-place BN + ReLU (+ optional FiLM)
// ------------------------------------------------------------------
__global__ void bnrelu_kernel(float* __restrict__ y, const float* __restrict__ stat,
                              const float* __restrict__ g, const float* __restrict__ bt,
                              int C, int P,
                              const float* __restrict__ scale, const float* __restrict__ shift) {
  size_t i = (size_t)blockIdx.x * blockDim.x + threadIdx.x;
  size_t total = (size_t)BATCH * C * P;
  if (i >= total) return;
  int c = (int)((i / (size_t)P) % (size_t)C);
  float v = y[i];
  v = (v - stat[c]) * stat[C + c] * g[c] + bt[c];
  v = fmaxf(v, 0.f);
  if (scale) v = v * (1.f + scale[i]) + shift[i];
  y[i] = v;
}

// ------------------------------------------------------------------
// boundary head layer 2: sigmoid(w2 . relu(bn(bdy1)) + b2) -> [B,1,N]
// ------------------------------------------------------------------
__global__ void bdy_kernel(const float* __restrict__ bdy1, const float* __restrict__ stat,
                           const float* __restrict__ g, const float* __restrict__ bt,
                           const float* __restrict__ w2, const float* __restrict__ b2,
                           float* __restrict__ outp) {
  const int b = blockIdx.y;
  const int n = blockIdx.x * blockDim.x + threadIdx.x;
  if (n >= NHR) return;
  const float* xb = bdy1 + (size_t)b * 32 * NHR;
  float acc = b2[0];
  #pragma unroll
  for (int c = 0; c < 32; ++c) {
    float v = xb[(size_t)c * NHR + n];
    v = (v - stat[c]) * stat[32 + c] * g[c] + bt[c];
    v = fmaxf(v, 0.f);
    acc = fmaf(w2[c], v, acc);
  }
  outp[(size_t)b * NHR + n] = 1.f / (1.f + expf(-acc));
}

// ------------------------------------------------------------------
// kNN: per hr point, top-16 nearest lr points; all 4096 lr xyz in LDS
// (48KB of the 320KB WGP pool). Also accumulates the 9 raw moments of
// rel_pos for the analytic BN2d stats of pos-enc layer 1.
// ------------------------------------------------------------------
__global__ void knn_kernel(const float* __restrict__ xyz_hr, const float* __restrict__ xyz_lr,
                           int* __restrict__ idx, float* __restrict__ mom) {
  __shared__ float sx[MLR];
  __shared__ float sy[MLR];
  __shared__ float sz[MLR];
  __shared__ float sMom[9];
  const int b = blockIdx.y;
  const float* lr = xyz_lr + (size_t)b * 3 * MLR;
  for (int i = threadIdx.x; i < MLR; i += blockDim.x) {
    sx[i] = lr[i];
    sy[i] = lr[MLR + i];
    sz[i] = lr[2 * MLR + i];
  }
  if (threadIdx.x < 9) sMom[threadIdx.x] = 0.f;
  __syncthreads();

  const int n = blockIdx.x * blockDim.x + threadIdx.x;
  const float* hr = xyz_hr + (size_t)b * 3 * NHR;
  const float px = hr[n], py = hr[NHR + n], pz = hr[2 * NHR + n];

  float bd[KNN];
  int   bi[KNN];
  #pragma unroll
  for (int k = 0; k < KNN; ++k) { bd[k] = 1e30f; bi[k] = 0; }
  float worst = 1e30f; int wslot = 0;

  for (int m = 0; m < MLR; ++m) {
    float dx = px - sx[m], dy = py - sy[m], dz = pz - sz[m];
    float d2 = fmaf(dx, dx, fmaf(dy, dy, dz * dz));
    if (d2 < worst) {
      bd[wslot] = d2; bi[wslot] = m;
      worst = -1.f;
      for (int k = 0; k < KNN; ++k)
        if (bd[k] > worst) { worst = bd[k]; wslot = k; }
    }
  }

  int* ib = idx + ((size_t)b * NHR + n) * KNN;
  float s[9] = {0.f, 0.f, 0.f, 0.f, 0.f, 0.f, 0.f, 0.f, 0.f};
  for (int k = 0; k < KNN; ++k) {
    int m = bi[k];
    ib[k] = m;
    float dx = px - sx[m], dy = py - sy[m], dz = pz - sz[m];
    s[0] += dx; s[1] += dy; s[2] += dz;
    s[3] += dx * dx; s[4] += dy * dy; s[5] += dz * dz;
    s[6] += dx * dy; s[7] += dx * dz; s[8] += dy * dz;
  }
  for (int j = 0; j < 9; ++j) {
    float v = s[j];
    #pragma unroll
    for (int off = 16; off > 0; off >>= 1) v += __shfl_xor(v, off);
    if ((threadIdx.x & 31) == 0) atomicAdd(&sMom[j], v);
  }
  __syncthreads();
  if (threadIdx.x < 9) atomicAdd(&mom[threadIdx.x], sMom[threadIdx.x]);
}

// ------------------------------------------------------------------
// analytic BN2d stats for pos-enc layer 1 (affine map of rel_pos)
// ------------------------------------------------------------------
__global__ void pe_finalize_kernel(const float* __restrict__ mom,
                                   const float* __restrict__ rp_w1, const float* __restrict__ rp_b1,
                                   float* __restrict__ peStat) {
  int c = threadIdx.x;
  if (c >= 64) return;
  const float cnt = (float)((size_t)BATCH * NHR * KNN);
  float mu0 = mom[0] / cnt, mu1 = mom[1] / cnt, mu2 = mom[2] / cnt;
  float E00 = mom[3] / cnt, E11 = mom[4] / cnt, E22 = mom[5] / cnt;
  float E01 = mom[6] / cnt, E02 = mom[7] / cnt, E12 = mom[8] / cnt;
  float w0 = rp_w1[c * 3], w1 = rp_w1[c * 3 + 1], w2 = rp_w1[c * 3 + 2];
  float bb = rp_b1[c];
  float wmu = w0 * mu0 + w1 * mu1 + w2 * mu2;
  float mean = wmu + bb;
  float E2 = w0 * w0 * E00 + w1 * w1 * E11 + w2 * w2 * E22
           + 2.f * (w0 * w1 * E01 + w0 * w2 * E02 + w1 * w2 * E12)
           + 2.f * bb * wmu + bb * bb;
  float var = E2 - mean * mean;
  peStat[c]      = mean;
  peStat[64 + c] = rsqrtf(var + EPSBN);
}

// ------------------------------------------------------------------
// WMMA attention: one wave per hr point. Each lane owns neighbor
// k = lane%16; PE1 (BN-folded, 3->64) is built directly in the WMMA
// B-fragment registers; pos = W2[64,64] @ PE1[64,16] is 8 WMMAs.
// Scores reduce over channels per-lane + one cross-half shfl; softmax
// and the 6-channel val gather reduce across the 16 neighbor lanes.
// ------------------------------------------------------------------
__global__ void attn_wmma_kernel(const float* __restrict__ xyz_hr, const float* __restrict__ xyz_lr,
                                 const float* __restrict__ val_lr,
                                 const float* __restrict__ Q, const float* __restrict__ Kf,
                                 const int* __restrict__ idx,
                                 const float* __restrict__ rp_w1, const float* __restrict__ rp_b1,
                                 const float* __restrict__ peStat,
                                 const float* __restrict__ rp_g, const float* __restrict__ rp_bt,
                                 const float* __restrict__ rp_w2, const float* __restrict__ rp_b2,
                                 float* __restrict__ out) {
  __shared__ _Float16 sW2[64 * 64]; // A-matrix (row m, K j) as f16
  __shared__ float sW1[64 * 3];
  __shared__ float sA[64];    // invstd * gamma
  __shared__ float sSh[64];   // (b1 - mean) * a + beta
  __shared__ float sB2[64];
  for (int i = threadIdx.x; i < 64 * 64; i += blockDim.x) sW2[i] = (_Float16)rp_w2[i];
  for (int i = threadIdx.x; i < 64 * 3; i += blockDim.x) sW1[i] = rp_w1[i];
  for (int i = threadIdx.x; i < 64; i += blockDim.x) {
    float a = peStat[64 + i] * rp_g[i];
    sA[i]  = a;
    sSh[i] = (rp_b1[i] - peStat[i]) * a + rp_bt[i];
    sB2[i] = rp_b2[i];
  }
  __syncthreads();

  const int wave = threadIdx.x >> 5;
  const int lane = threadIdx.x & 31;
  const int lcol = lane & 15;   // neighbor index k
  const int lhi  = lane >> 4;

  const int b = blockIdx.y;
  const int n = blockIdx.x * 8 + wave;
  if (n >= NHR) return;

  const float* Qb = Q + (size_t)b * 64 * NHR;
  const float* Kb = Kf + (size_t)b * 64 * MLR;
  const float* lr = xyz_lr + (size_t)b * 3 * MLR;
  const float* hr = xyz_hr + (size_t)b * 3 * NHR;
  const float* vb = val_lr + (size_t)b * 6 * MLR;
  const float px = hr[n], py = hr[NHR + n], pz = hr[2 * NHR + n];

  const int ml = idx[((size_t)b * NHR + n) * KNN + lcol]; // my neighbor
  __builtin_prefetch(Kb + ml, 0, 1);
  const float rx = px - lr[ml], ry = py - lr[MLR + ml], rz = pz - lr[2 * MLR + ml];

  v8f acc[4] = {};
  #pragma unroll
  for (int ks = 0; ks < 2; ++ks) {
    const int k0 = ks * 32;
    // B fragment: col = neighbor lcol, K(channel j) = k0 + lhi*16 + h
    v16h bx;
    #pragma unroll
    for (int h = 0; h < 16; ++h) {
      const int j = k0 + lhi * 16 + h;
      float raw = fmaf(sW1[j * 3], rx, fmaf(sW1[j * 3 + 1], ry, sW1[j * 3 + 2] * rz));
      float v = fmaxf(fmaf(raw, sA[j], sSh[j]), 0.f);
      bx[h] = (_Float16)v;
    }
    #pragma unroll
    for (int mt = 0; mt < 4; ++mt) {
      const int m = mt * 16 + lcol;
      v16h ax;
      #pragma unroll
      for (int h = 0; h < 8; ++h)
        ax[h] = sW2[m * 64 + k0 + lhi * 16 + h];
      #pragma unroll
      for (int h = 0; h < 8; ++h)
        ax[8 + h] = sW2[m * 64 + k0 + lhi * 16 + 8 + h];
      acc[mt] = __builtin_amdgcn_wmma_f32_16x16x32_f16(
          false, ax, false, bx, (short)0, acc[mt], false, false);
    }
  }

  // score partial: this lane holds rows m = mt*16 + lhi*8 + r, col = lcol
  float ssum = 0.f;
  #pragma unroll
  for (int mt = 0; mt < 4; ++mt) {
    #pragma unroll
    for (int r = 0; r < 8; ++r) {
      const int m = mt * 16 + lhi * 8 + r;
      float pos = acc[mt][r] + sB2[m];
      float kg = Kb[(size_t)m * MLR + ml];
      ssum = fmaf(Qb[(size_t)m * NHR + n], kg + pos, ssum);
    }
  }
  ssum += __shfl_xor(ssum, 16);      // combine the two channel halves
  float score = ssum * 0.125f;       // 1/sqrt(64)

  // softmax over the 16 neighbor lanes (both halves identical)
  float mx = score;
  #pragma unroll
  for (int off = 1; off < 16; off <<= 1) mx = fmaxf(mx, __shfl_xor(mx, off));
  float e = expf(score - mx);
  float den = e;
  #pragma unroll
  for (int off = 1; off < 16; off <<= 1) den += __shfl_xor(den, off);
  const float aw = e / den;

  // weighted gather of 6-channel values, reduce across neighbor lanes
  #pragma unroll
  for (int cv = 0; cv < 6; ++cv) {
    float pv = aw * vb[(size_t)cv * MLR + ml];
    #pragma unroll
    for (int off = 1; off < 16; off <<= 1) pv += __shfl_xor(pv, off);
    if (lane == 0) out[(size_t)b * 6 * NHR + (size_t)cv * NHR + n] = pv;
  }
}

// ------------------------------------------------------------------
// host launch
// ------------------------------------------------------------------
extern "C" void kernel_launch(void* const* d_in, const int* in_sizes, int n_in,
                              void* d_out, int out_size, void* d_ws, size_t ws_size,
                              hipStream_t stream) {
  const float* xyz_hr  = (const float*)d_in[0];
  const float* xyz_lr  = (const float*)d_in[1];
  const float* val_lr  = (const float*)d_in[2];
  const float* feat_hr = (const float*)d_in[3];
  const float* feat_lr = (const float*)d_in[4];
  const float* ge_w1 = (const float*)d_in[5];
  const float* ge_b1 = (const float*)d_in[6];
  const float* ge_g1 = (const float*)d_in[7];
  const float* ge_bt1= (const float*)d_in[8];
  const float* ge_w2 = (const float*)d_in[9];
  const float* ge_b2 = (const float*)d_in[10];
  const float* ge_g2 = (const float*)d_in[11];
  const float* ge_bt2= (const float*)d_in[12];
  const float* sc_w  = (const float*)d_in[13];
  const float* sc_b  = (const float*)d_in[14];
  const float* sh_w  = (const float*)d_in[15];
  const float* sh_b  = (const float*)d_in[16];
  const float* q_w   = (const float*)d_in[17];
  const float* q_b   = (const float*)d_in[18];
  const float* k_w   = (const float*)d_in[19];
  const float* k_b   = (const float*)d_in[20];
  const float* bh_w1 = (const float*)d_in[21];
  const float* bh_b1 = (const float*)d_in[22];
  const float* bh_g  = (const float*)d_in[23];
  const float* bh_bt = (const float*)d_in[24];
  const float* bh_w2 = (const float*)d_in[25];
  const float* bh_b2 = (const float*)d_in[26];
  const float* rp_w1 = (const float*)d_in[27];
  const float* rp_b1 = (const float*)d_in[28];
  const float* rp_g  = (const float*)d_in[29];
  const float* rp_bt = (const float*)d_in[30];
  const float* rp_w2 = (const float*)d_in[31];
  const float* rp_b2 = (const float*)d_in[32];

  float* ws = (float*)d_ws;
  // workspace layout (floats)
  const size_t G1HR  = 0;                       // B*64*N
  const size_t G2HR  = G1HR  + (size_t)BATCH * 64 * NHR;
  const size_t QBUF  = G2HR  + (size_t)BATCH * 64 * NHR;
  const size_t G1LR  = QBUF  + (size_t)BATCH * 64 * NHR;
  const size_t G2LR  = G1LR  + (size_t)BATCH * 64 * MLR;
  const size_t SCALE = G2LR  + (size_t)BATCH * 64 * MLR;
  const size_t SHIFT = SCALE + (size_t)BATCH * 64 * MLR;
  const size_t KFBUF = SHIFT + (size_t)BATCH * 64 * MLR;
  const size_t BDY1  = KFBUF + (size_t)BATCH * 64 * MLR;
  const size_t IDXB  = BDY1  + (size_t)BATCH * 32 * NHR;
  const size_t STATS = IDXB  + (size_t)BATCH * NHR * KNN; // idx as int, same footprint
  float* st_g1hr = ws + STATS;        // 128
  float* st_g1lr = st_g1hr + 128;     // 128
  float* st_g2hr = st_g1lr + 128;     // 128
  float* st_g2lr = st_g2hr + 128;     // 128
  float* st_bdy  = st_g2lr + 128;     // 64
  float* momb    = st_bdy + 64;       // 9
  float* peStat  = momb + 9;          // 128
  const int statN = 128 * 4 + 64 + 9 + 128;
  int* idxbuf = (int*)(ws + IDXB);

  float* outMain = (float*)d_out;                             // [B,6,N]
  float* outBdy  = outMain + (size_t)BATCH * 6 * NHR;         // [B,1,N]

  // 0) zero stats
  zero_kernel<<<(statN + 255) / 256, 256, 0, stream>>>(st_g1hr, statN);

  // 1) geom layer 1 (18->64), stats
  conv_small_kernel<GEOC, 64, true><<<dim3(NHR / 256, BATCH), 256, 0, stream>>>(
      feat_hr, ge_w1, ge_b1, ws + G1HR, NHR, st_g1hr);
  conv_small_kernel<GEOC, 64, true><<<dim3(MLR / 256, BATCH), 256, 0, stream>>>(
      feat_lr, ge_w1, ge_b1, ws + G1LR, MLR, st_g1lr);
  bn_finalize_kernel<<<1, 64, 0, stream>>>(st_g1hr, 64, (float)(BATCH * NHR));
  bn_finalize_kernel<<<1, 64, 0, stream>>>(st_g1lr, 64, (float)(BATCH * MLR));
  bnrelu_kernel<<<(BATCH * 64 * NHR) / 256, 256, 0, stream>>>(
      ws + G1HR, st_g1hr, ge_g1, ge_bt1, 64, NHR, nullptr, nullptr);
  bnrelu_kernel<<<(BATCH * 64 * MLR) / 256, 256, 0, stream>>>(
      ws + G1LR, st_g1lr, ge_g1, ge_bt1, 64, MLR, nullptr, nullptr);

  // 2) geom layer 2 (64->64) via WMMA, stats
  conv64_wmma_kernel<<<dim3(NHR / 128, BATCH), 256, 0, stream>>>(
      ws + G1HR, ge_w2, ge_b2, ws + G2HR, NHR, st_g2hr);
  conv64_wmma_kernel<<<dim3(MLR / 128, BATCH), 256, 0, stream>>>(
      ws + G1LR, ge_w2, ge_b2, ws + G2LR, MLR, st_g2lr);
  bn_finalize_kernel<<<1, 64, 0, stream>>>(st_g2hr, 64, (float)(BATCH * NHR));
  bn_finalize_kernel<<<1, 64, 0, stream>>>(st_g2lr, 64, (float)(BATCH * MLR));

  // 3) FiLM scale/shift (6->64) for lr
  conv_small_kernel<6, 64, false><<<dim3(MLR / 256, BATCH), 256, 0, stream>>>(
      val_lr, sc_w, sc_b, ws + SCALE, MLR, nullptr);
  conv_small_kernel<6, 64, false><<<dim3(MLR / 256, BATCH), 256, 0, stream>>>(
      val_lr, sh_w, sh_b, ws + SHIFT, MLR, nullptr);

  // 4) BN+ReLU (+FiLM on lr) -> geom_hr / filmed geom_lr, in place
  bnrelu_kernel<<<(BATCH * 64 * NHR) / 256, 256, 0, stream>>>(
      ws + G2HR, st_g2hr, ge_g2, ge_bt2, 64, NHR, nullptr, nullptr);
  bnrelu_kernel<<<(BATCH * 64 * MLR) / 256, 256, 0, stream>>>(
      ws + G2LR, st_g2lr, ge_g2, ge_bt2, 64, MLR, ws + SCALE, ws + SHIFT);

  // 5) boundary head
  conv_small_kernel<64, 32, true><<<dim3(NHR / 256, BATCH), 256, 0, stream>>>(
      ws + G2HR, bh_w1, bh_b1, ws + BDY1, NHR, st_bdy);
  bn_finalize_kernel<<<1, 64, 0, stream>>>(st_bdy, 32, (float)(BATCH * NHR));
  bdy_kernel<<<dim3(NHR / 256, BATCH), 256, 0, stream>>>(
      ws + BDY1, st_bdy, bh_g, bh_bt, bh_w2, bh_b2, outBdy);

  // 6) Q / K projections via WMMA
  conv64_wmma_kernel<<<dim3(NHR / 128, BATCH), 256, 0, stream>>>(
      ws + G2HR, q_w, q_b, ws + QBUF, NHR, nullptr);
  conv64_wmma_kernel<<<dim3(MLR / 128, BATCH), 256, 0, stream>>>(
      ws + G2LR, k_w, k_b, ws + KFBUF, MLR, nullptr);

  // 7) kNN + rel_pos moments
  knn_kernel<<<dim3(NHR / 256, BATCH), 256, 0, stream>>>(xyz_hr, xyz_lr, idxbuf, momb);
  pe_finalize_kernel<<<1, 64, 0, stream>>>(momb, rp_w1, rp_b1, peStat);

  // 8) attention (WMMA pos-enc) + output; one wave per point
  attn_wmma_kernel<<<dim3(NHR / 8, BATCH), 256, 0, stream>>>(
      xyz_hr, xyz_lr, val_lr, ws + QBUF, ws + KFBUF, idxbuf,
      rp_w1, rp_b1, peStat, rp_g, rp_bt, rp_w2, rp_b2, outMain);
}
